// TinyGPT_74904229642817
// MI455X (gfx1250) — compile-verified, matrix operand order checked by reading, not run on hardware
//
#include <hip/hip_runtime.h>
#include <hip/hip_bf16.h>

// ---------------- model constants ----------------
#define T_    2048
#define E_    1024
#define NH    16
#define HD    64
#define FF    4096
#define NL    8
#define VV    32000
#define C1_   2000
#define C2_   10000
#define NH1   256      // E/4
#define NH2   64       // E/16
#define HEADN 2002     // C1+2

// ---------------- WMMA GEMM tiling ----------------
#define BM   128
#define BN   128
#define BK   32
#define LSTR 40        // padded LDS row stride in bf16 elems (80B, 16B aligned)

typedef __attribute__((ext_vector_type(16))) __bf16 v16bf;
typedef __attribute__((ext_vector_type(8)))  float  v8f;

enum { EPI_NONE = 0, EPI_GELU = 1, EPI_RES = 2, EPI_CAUSAL = 3 };

__device__ __forceinline__ unsigned short f2bf(float f) {
  union { float f; unsigned u; } x; x.f = f;
  unsigned r = x.u + 0x7FFFu + ((x.u >> 16) & 1u);   // round-to-nearest-even
  return (unsigned short)(r >> 16);
}

// CDNA5 async global->LDS copy, 16B per lane, tracked by ASYNCcnt.
__device__ __forceinline__ void async_copy_b128(unsigned lds_byte_addr,
                                                const void* gptr) {
  asm volatile("global_load_async_to_lds_b128 %0, %1, off"
               :: "v"(lds_byte_addr), "v"(gptr) : "memory");
}

// C[M,N](f32 or bf16) = A[M,K](bf16,row-major) * Bt[N,K](bf16,row-major) + epilogue
// Requires: M % 128 == 0, K % 32 == 0. N arbitrary (guarded).
template <int EPI>
__global__ __launch_bounds__(256)
void gemm_bf16_wmma(const unsigned short* __restrict__ A, int lda,
                    const unsigned short* __restrict__ Bt, int ldb,
                    void* __restrict__ Cv, int ldc, int outBF16,
                    const float* __restrict__ bias,
                    const float* __restrict__ res, int ldres,
                    int M, int N, int K, float alpha) {
  __shared__ unsigned short lA[2][BM * LSTR];
  __shared__ unsigned short lB[2][BN * LSTR];

  const int tid  = threadIdx.x;
  const int m0   = blockIdx.y * BM;
  const int n0   = blockIdx.x * BN;
  const int lane = tid & 31;
  const int wid  = tid >> 5;          // 8 waves
  const int wm   = wid & 3;           // 4 waves along M (32 rows each)
  const int wn   = wid >> 2;          // 2 waves along N (64 cols each)
  const int half = lane >> 4;         // lane group (0: lanes 0-15, 1: 16-31)
  const int lm   = lane & 15;
  const bool fullN = (n0 + BN) <= N;  // uniform per block

  v8f acc[2][4];
  const v8f vzero = {0.f, 0.f, 0.f, 0.f, 0.f, 0.f, 0.f, 0.f};
#pragma unroll
  for (int i = 0; i < 2; ++i)
#pragma unroll
    for (int j = 0; j < 4; ++j) acc[i][j] = vzero;

  // Stage one 128x32 A tile and one 128x32 Bt tile into LDS buffer `buf`.
  // Full tiles: CDNA5 async copies (4 per thread -> ASYNCcnt += 4 per wave).
  // Partial-N tiles: synchronous path with zero fill (keeps ASYNCcnt uniform).
  auto stage = [&](int buf, int k0) {
    if (fullN) {
#pragma unroll
      for (int p = 0; p < 2; ++p) {
        const int idx = (p * 256 + tid) * 8;   // 0..4095
        const int r   = idx >> 5;              // /32
        const int c   = idx & 31;
        async_copy_b128((unsigned)(size_t)&lA[buf][r * LSTR + c],
                        A + (size_t)(m0 + r) * lda + k0 + c);
        async_copy_b128((unsigned)(size_t)&lB[buf][r * LSTR + c],
                        Bt + (size_t)(n0 + r) * ldb + k0 + c);
      }
    } else {
#pragma unroll
      for (int p = 0; p < 2; ++p) {
        const int idx = (p * 256 + tid) * 8;
        const int r   = idx >> 5;
        const int c   = idx & 31;
        uint4 av = *reinterpret_cast<const uint4*>(A + (size_t)(m0 + r) * lda + k0 + c);
        *reinterpret_cast<uint4*>(&lA[buf][r * LSTR + c]) = av;
        uint4 bv = make_uint4(0u, 0u, 0u, 0u);
        if (n0 + r < N)
          bv = *reinterpret_cast<const uint4*>(Bt + (size_t)(n0 + r) * ldb + k0 + c);
        *reinterpret_cast<uint4*>(&lB[buf][r * LSTR + c]) = bv;
      }
    }
  };

  const int kTiles = K / BK;
  stage(0, 0);                                   // prologue
  for (int kt = 0; kt < kTiles; ++kt) {
    const bool hasNext = (kt + 1) < kTiles;
    if (hasNext) stage((kt + 1) & 1, (kt + 1) * BK);   // overlap next stage
    if (fullN) {
      // in-order completion: <=4 outstanding => stage kt fully landed in LDS
      if (hasNext) asm volatile("s_wait_asynccnt 0x4" ::: "memory");
      else         asm volatile("s_wait_asynccnt 0x0" ::: "memory");
    }
    __syncthreads();

    const unsigned short* bufA = lA[kt & 1];
    const unsigned short* bufB = lB[kt & 1];
    const int kb = half * 8;                     // per-lane K chunk base
    union Frag { uint4 q[2]; v16bf v; } af[2], bfr[4];
#pragma unroll
    for (int mi = 0; mi < 2; ++mi) {
      const unsigned short* p0 = &bufA[(wm * 32 + mi * 16 + lm) * LSTR + kb];
      af[mi].q[0] = *reinterpret_cast<const uint4*>(p0);
      af[mi].q[1] = *reinterpret_cast<const uint4*>(p0 + 16);
    }
#pragma unroll
    for (int ni = 0; ni < 4; ++ni) {
      const unsigned short* p0 = &bufB[(wn * 64 + ni * 16 + lm) * LSTR + kb];
      bfr[ni].q[0] = *reinterpret_cast<const uint4*>(p0);
      bfr[ni].q[1] = *reinterpret_cast<const uint4*>(p0 + 16);
    }
#pragma unroll
    for (int mi = 0; mi < 2; ++mi)
#pragma unroll
      for (int ni = 0; ni < 4; ++ni)
        acc[mi][ni] = __builtin_amdgcn_wmma_f32_16x16x32_bf16(
            false, af[mi].v, false, bfr[ni].v, (short)0, acc[mi][ni], false, false);
    __syncthreads();
  }

  // ---- epilogue + store (C layout: row = +half*8 + r, col = lane&15) ----
#pragma unroll
  for (int mi = 0; mi < 2; ++mi) {
#pragma unroll
    for (int ni = 0; ni < 4; ++ni) {
      const int col = n0 + wn * 64 + ni * 16 + lm;
      if (col >= N) continue;
      const float bv = (EPI != EPI_CAUSAL && bias) ? bias[col] : 0.f;
#pragma unroll
      for (int r = 0; r < 8; ++r) {
        const int row = m0 + wm * 32 + mi * 16 + half * 8 + r;
        float v = acc[mi][ni][r];
        if (EPI == EPI_CAUSAL) {
          v *= alpha;
          if (col > row) v = -1e30f;           // causal mask (query=row, key=col)
        } else {
          v += bv;
          if (EPI == EPI_GELU) v = 0.5f * v * (1.f + erff(v * 0.70710678118f));
          if (EPI == EPI_RES)  v += res[(size_t)row * ldres + col];
        }
        if (outBF16) ((unsigned short*)Cv)[(size_t)row * ldc + col] = f2bf(v);
        else         ((float*)Cv)[(size_t)row * ldc + col] = v;
      }
    }
  }
}

// ---------------- elementwise / row kernels ----------------
__global__ __launch_bounds__(256)
void embed_kernel(const int* __restrict__ x, const float* __restrict__ tok,
                  const float* __restrict__ pos, float* __restrict__ h) {
  const int t = blockIdx.x, tid = threadIdx.x;
  const int id = x[t];
#pragma unroll
  for (int i = 0; i < 4; ++i) {
    const int e = tid + i * 256;
    h[(size_t)t * E_ + e] = tok[(size_t)id * E_ + e] + pos[(size_t)t * E_ + e];
  }
}

// f32 [K,N] -> bf16 [N,K] (transpose + convert)
__global__ __launch_bounds__(256)
void convT_kernel(const float* __restrict__ w, unsigned short* __restrict__ wt,
                  int K, int N) {
  const long long i = (long long)blockIdx.x * 256 + threadIdx.x;
  if (i >= (long long)K * N) return;
  const int n = (int)(i / K), k = (int)(i % K);
  wt[(size_t)n * K + k] = f2bf(w[(size_t)k * N + n]);
}

// LayerNorm over E=1024, output bf16 (GEMM A operand)
__global__ __launch_bounds__(256)
void ln_bf16_kernel(const float* __restrict__ x, const float* __restrict__ w,
                    const float* __restrict__ b, unsigned short* __restrict__ y) {
  __shared__ float s1[256], s2[256];
  const int row = blockIdx.x, tid = threadIdx.x;
  const float* xr = x + (size_t)row * E_;
  float lsum = 0.f, lsq = 0.f, vals[4];
#pragma unroll
  for (int i = 0; i < 4; ++i) {
    const float v = xr[tid + i * 256];
    vals[i] = v; lsum += v; lsq += v * v;
  }
  s1[tid] = lsum; s2[tid] = lsq; __syncthreads();
  for (int s = 128; s > 0; s >>= 1) {
    if (tid < s) { s1[tid] += s1[tid + s]; s2[tid] += s2[tid + s]; }
    __syncthreads();
  }
  const float mean = s1[0] * (1.f / E_);
  const float var  = s2[0] * (1.f / E_) - mean * mean;
  const float rstd = rsqrtf(var + 1e-5f);
#pragma unroll
  for (int i = 0; i < 4; ++i) {
    const int e = tid + i * 256;
    y[(size_t)row * E_ + e] = f2bf((vals[i] - mean) * rstd * w[e] + b[e]);
  }
}

// V slice of qkv (bf16 [T, 3E]) -> Vt bf16 [E, T]
__global__ __launch_bounds__(256)
void vtrans_kernel(const unsigned short* __restrict__ qkvb,
                   unsigned short* __restrict__ vtb) {
  const int i = blockIdx.x * 256 + threadIdx.x;   // < T*E
  const int t = i >> 10, c = i & 1023;
  vtb[(size_t)c * T_ + t] = qkvb[(size_t)t * (3 * E_) + 2 * E_ + c];
}

// attention row softmax: S f32 [T,T] -> P bf16 [T,T]
__global__ __launch_bounds__(256)
void softmax_kernel(const float* __restrict__ S, unsigned short* __restrict__ P) {
  __shared__ float red[256];
  const int row = blockIdx.x, tid = threadIdx.x;
  const float* sr = S + (size_t)row * T_;
  float m = -1e30f;
  for (int c = tid; c < T_; c += 256) m = fmaxf(m, sr[c]);
  red[tid] = m; __syncthreads();
  for (int s = 128; s > 0; s >>= 1) { if (tid < s) red[tid] = fmaxf(red[tid], red[tid + s]); __syncthreads(); }
  m = red[0]; __syncthreads();
  float sum = 0.f;
  for (int c = tid; c < T_; c += 256) sum += __expf(sr[c] - m);
  red[tid] = sum; __syncthreads();
  for (int s = 128; s > 0; s >>= 1) { if (tid < s) red[tid] += red[tid + s]; __syncthreads(); }
  const float inv = 1.f / red[0];
  for (int c = tid; c < T_; c += 256) P[(size_t)row * T_ + c] = f2bf(__expf(sr[c] - m) * inv);
}

// head log-softmax: logits [T,2002] -> out[:, :2000], offs[row] = lp[2000],lp[2001]
__global__ __launch_bounds__(256)
void head_lsm_kernel(const float* __restrict__ hl, float* __restrict__ out,
                     float* __restrict__ offs) {
  __shared__ float red[256];
  const int row = blockIdx.x, tid = threadIdx.x;
  const float* r_ = hl + (size_t)row * HEADN;
  float m = -1e30f;
  for (int c = tid; c < HEADN; c += 256) m = fmaxf(m, r_[c]);
  red[tid] = m; __syncthreads();
  for (int s = 128; s > 0; s >>= 1) { if (tid < s) red[tid] = fmaxf(red[tid], red[tid + s]); __syncthreads(); }
  m = red[0]; __syncthreads();
  float sum = 0.f;
  for (int c = tid; c < HEADN; c += 256) sum += expf(r_[c] - m);
  red[tid] = sum; __syncthreads();
  for (int s = 128; s > 0; s >>= 1) { if (tid < s) red[tid] += red[tid + s]; __syncthreads(); }
  const float lse = m + logf(red[0]);
  for (int c = tid; c < HEADN; c += 256) {
    const float lp = r_[c] - lse;
    if (c < C1_) out[(size_t)row * VV + c] = lp;
    else         offs[row * 2 + (c - C1_)] = lp;
  }
}

// in-place tail log-softmax on d_out slice + head cluster offset
__global__ __launch_bounds__(256)
void tail_lsm_kernel(float* __restrict__ base, int NC, int coloff,
                     const float* __restrict__ offs, int which) {
  __shared__ float red[256];
  const int row = blockIdx.x, tid = threadIdx.x;
  float* r_ = base + (size_t)row * VV + coloff;
  float m = -1e30f;
  for (int c = tid; c < NC; c += 256) m = fmaxf(m, r_[c]);
  red[tid] = m; __syncthreads();
  for (int s = 128; s > 0; s >>= 1) { if (tid < s) red[tid] = fmaxf(red[tid], red[tid + s]); __syncthreads(); }
  m = red[0]; __syncthreads();
  float sum = 0.f;
  for (int c = tid; c < NC; c += 256) sum += expf(r_[c] - m);
  red[tid] = sum; __syncthreads();
  for (int s = 128; s > 0; s >>= 1) { if (tid < s) red[tid] += red[tid + s]; __syncthreads(); }
  const float lse = m + logf(red[0]);
  const float add = offs[row * 2 + which];
  for (int c = tid; c < NC; c += 256) r_[c] = r_[c] - lse + add;
}

// ---------------- host orchestration ----------------
extern "C" void kernel_launch(void* const* d_in, const int* in_sizes, int n_in,
                              void* d_out, int out_size, void* d_ws, size_t ws_size,
                              hipStream_t stream) {
  (void)in_sizes; (void)n_in; (void)out_size; (void)ws_size;
  const int*   x       = (const int*)  d_in[0];
  const float* tok_emb = (const float*)d_in[1];
  const float* pos_emb = (const float*)d_in[2];
  const float* ln1_w   = (const float*)d_in[3];
  const float* ln1_b   = (const float*)d_in[4];
  const float* qkv_w   = (const float*)d_in[5];
  const float* qkv_b   = (const float*)d_in[6];
  const float* out_w   = (const float*)d_in[7];
  const float* out_b   = (const float*)d_in[8];
  const float* ln2_w   = (const float*)d_in[9];
  const float* ln2_b   = (const float*)d_in[10];
  const float* ff1_w   = (const float*)d_in[11];
  const float* ff1_b   = (const float*)d_in[12];
  const float* ff2_w   = (const float*)d_in[13];
  const float* ff2_b   = (const float*)d_in[14];
  const float* lnf_w   = (const float*)d_in[15];
  const float* lnf_b   = (const float*)d_in[16];
  const float* head_w  = (const float*)d_in[17];
  const float* tail0_p = (const float*)d_in[18];
  const float* tail0_o = (const float*)d_in[19];
  const float* tail1_p = (const float*)d_in[20];
  const float* tail1_o = (const float*)d_in[21];
  float* out = (float*)d_out;

  char* wsb = (char*)d_ws;
  size_t off = 0;
  auto carve = [&](size_t bytes) -> void* {
    void* p = wsb + off;
    off += (bytes + 255) & ~(size_t)255;
    return p;
  };
  typedef unsigned short us;
  us* wq_t    = (us*)carve((size_t)NL * 3 * E_ * E_ * 2);
  us* wo_t    = (us*)carve((size_t)NL * E_ * E_ * 2);
  us* wf1_t   = (us*)carve((size_t)NL * FF * E_ * 2);
  us* wf2_t   = (us*)carve((size_t)NL * E_ * FF * 2);
  us* whead_t = (us*)carve((size_t)HEADN * E_ * 2);
  us* wt0p_t  = (us*)carve((size_t)NH1 * E_ * 2);
  us* wt0o_t  = (us*)carve((size_t)(C2_ - C1_) * NH1 * 2);
  us* wt1p_t  = (us*)carve((size_t)NH2 * E_ * 2);
  us* wt1o_t  = (us*)carve((size_t)(VV - C2_) * NH2 * 2);
  float* h    = (float*)carve((size_t)T_ * E_ * 4);
  us* ybf     = (us*)carve((size_t)T_ * E_ * 2);
  us* qkvb    = (us*)carve((size_t)T_ * 3 * E_ * 2);
  us* vtb     = (us*)carve((size_t)E_ * T_ * 2);
  float* S    = (float*)carve((size_t)T_ * T_ * 4);
  us* Pb      = (us*)carve((size_t)T_ * T_ * 2);
  us* attnOb  = (us*)carve((size_t)T_ * E_ * 2);
  us* ffhb    = (us*)carve((size_t)T_ * FF * 2);
  float* hl   = (float*)carve((size_t)T_ * HEADN * 4);
  float* offs = (float*)carve((size_t)T_ * 2 * 4);
  us* t0hb    = (us*)carve((size_t)T_ * NH1 * 2);
  us* t1hb    = (us*)carve((size_t)T_ * NH2 * 2);

  auto convT = [&](const float* w, us* wt, int K, int N) {
    const long long tot = (long long)K * N;
    convT_kernel<<<(unsigned)((tot + 255) / 256), 256, 0, stream>>>(w, wt, K, N);
  };
  auto gemm = [&](int epi, const us* A, int lda, const us* Bt, int ldb,
                  void* C, int ldc, int obf, const float* bias,
                  const float* res, int ldres, int M, int N, int K, float alpha) {
    dim3 g((N + BN - 1) / BN, M / BM);
    switch (epi) {
      case EPI_NONE:   gemm_bf16_wmma<EPI_NONE>  <<<g, 256, 0, stream>>>(A, lda, Bt, ldb, C, ldc, obf, bias, res, ldres, M, N, K, alpha); break;
      case EPI_GELU:   gemm_bf16_wmma<EPI_GELU>  <<<g, 256, 0, stream>>>(A, lda, Bt, ldb, C, ldc, obf, bias, res, ldres, M, N, K, alpha); break;
      case EPI_RES:    gemm_bf16_wmma<EPI_RES>   <<<g, 256, 0, stream>>>(A, lda, Bt, ldb, C, ldc, obf, bias, res, ldres, M, N, K, alpha); break;
      default:         gemm_bf16_wmma<EPI_CAUSAL><<<g, 256, 0, stream>>>(A, lda, Bt, ldb, C, ldc, obf, bias, res, ldres, M, N, K, alpha); break;
    }
  };

  // ---- weight conversion (f32 [K,N] -> bf16 [N,K]) ----
  for (int l = 0; l < NL; ++l) {
    convT(qkv_w + (size_t)l * E_ * 3 * E_, wq_t  + (size_t)l * 3 * E_ * E_, E_, 3 * E_);
    convT(out_w + (size_t)l * E_ * E_,     wo_t  + (size_t)l * E_ * E_,     E_, E_);
    convT(ff1_w + (size_t)l * E_ * FF,     wf1_t + (size_t)l * FF * E_,     E_, FF);
    convT(ff2_w + (size_t)l * FF * E_,     wf2_t + (size_t)l * E_ * FF,     FF, E_);
  }
  convT(head_w,  whead_t, E_,  HEADN);
  convT(tail0_p, wt0p_t,  E_,  NH1);
  convT(tail0_o, wt0o_t,  NH1, C2_ - C1_);
  convT(tail1_p, wt1p_t,  E_,  NH2);
  convT(tail1_o, wt1o_t,  NH2, VV - C2_);

  // ---- embedding ----
  embed_kernel<<<T_, 256, 0, stream>>>(x, tok_emb, pos_emb, h);

  // ---- transformer layers ----
  const float scale = 0.125f;  // 1/sqrt(64)
  for (int l = 0; l < NL; ++l) {
    ln_bf16_kernel<<<T_, 256, 0, stream>>>(h, ln1_w + l * E_, ln1_b + l * E_, ybf);
    // qkv projection -> bf16 [T, 3E]
    gemm(EPI_NONE, ybf, E_, wq_t + (size_t)l * 3 * E_ * E_, E_,
         qkvb, 3 * E_, 1, qkv_b + (size_t)l * 3 * E_, nullptr, 0, T_, 3 * E_, E_, 1.f);
    vtrans_kernel<<<(T_ * E_) / 256, 256, 0, stream>>>(qkvb, vtb);
    for (int hh = 0; hh < NH; ++hh) {
      // S = scale * Q K^T with causal mask (Q,K read in place from qkv buffer)
      gemm(EPI_CAUSAL, qkvb + hh * HD, 3 * E_, qkvb + E_ + hh * HD, 3 * E_,
           S, T_, 0, nullptr, nullptr, 0, T_, T_, HD, scale);
      softmax_kernel<<<T_, 256, 0, stream>>>(S, Pb);
      // O_h = P @ V_h  (bf16 out into per-head columns)
      gemm(EPI_NONE, Pb, T_, vtb + (size_t)hh * HD * T_, T_,
           attnOb + hh * HD, E_, 1, nullptr, nullptr, 0, T_, HD, T_, 1.f);
    }
    // out projection + residual
    gemm(EPI_RES, attnOb, E_, wo_t + (size_t)l * E_ * E_, E_,
         h, E_, 0, out_b + (size_t)l * E_, h, E_, T_, E_, E_, 1.f);
    ln_bf16_kernel<<<T_, 256, 0, stream>>>(h, ln2_w + l * E_, ln2_b + l * E_, ybf);
    // FF1 + exact GELU -> bf16
    gemm(EPI_GELU, ybf, E_, wf1_t + (size_t)l * FF * E_, E_,
         ffhb, FF, 1, ff1_b + (size_t)l * FF, nullptr, 0, T_, FF, E_, 1.f);
    // FF2 + residual
    gemm(EPI_RES, ffhb, FF, wf2_t + (size_t)l * E_ * FF, FF,
         h, E_, 0, ff2_b + (size_t)l * E_, h, E_, T_, E_, FF, 1.f);
  }

  // ---- final LN + adaptive log-softmax ----
  ln_bf16_kernel<<<T_, 256, 0, stream>>>(h, lnf_w, lnf_b, ybf);

  // shortlist head
  gemm(EPI_NONE, ybf, E_, whead_t, E_, hl, HEADN, 0, nullptr, nullptr, 0, T_, HEADN, E_, 1.f);
  head_lsm_kernel<<<T_, 256, 0, stream>>>(hl, out, offs);

  // tail cluster 0: E->256->8000, logits straight into d_out[:, 2000:10000]
  gemm(EPI_NONE, ybf, E_, wt0p_t, E_, t0hb, NH1, 1, nullptr, nullptr, 0, T_, NH1, E_, 1.f);
  gemm(EPI_NONE, t0hb, NH1, wt0o_t, NH1, out + C1_, VV, 0, nullptr, nullptr, 0, T_, C2_ - C1_, NH1, 1.f);
  tail_lsm_kernel<<<T_, 256, 0, stream>>>(out, C2_ - C1_, C1_, offs, 0);

  // tail cluster 1: E->64->22000, logits straight into d_out[:, 10000:32000]
  gemm(EPI_NONE, ybf, E_, wt1p_t, E_, t1hb, NH2, 1, nullptr, nullptr, 0, T_, NH2, E_, 1.f);
  gemm(EPI_NONE, t1hb, NH2, wt1o_t, NH2, out + C2_, VV, 0, nullptr, nullptr, 0, T_, VV - C2_, NH2, 1.f);
  tail_lsm_kernel<<<T_, 256, 0, stream>>>(out, VV - C2_, C2_, offs, 1);
}
